// Dev_LSTM_13340168421544
// MI455X (gfx1250) — matrix operand order repaired and checked by reading, not compiled, not run on hardware
//
#include <hip/hip_runtime.h>
#include <stdint.h>

// ---------- types ----------
typedef __attribute__((ext_vector_type(16))) __bf16 v16bf;
typedef __attribute__((ext_vector_type(8)))  float  v8f;
typedef __attribute__((ext_vector_type(4)))  unsigned int u32x4;
typedef __attribute__((ext_vector_type(4)))  float  f32x4;
typedef unsigned short ushort;

struct Q2 { u32x4 lo, hi; };
union Frag32B { Q2 q; v16bf v; };

// ---------- problem constants ----------
#define BB   64
#define SS   512
#define II   1024
#define HH   1024
#define G4   4096          // 4*H
#define KK   1024          // reduction dim (I or H)
#define MM   (BB*SS)       // 32768 rows of x

// ---------- helpers ----------
__device__ __forceinline__ ushort f2bf(float f) {
  union { float f; unsigned int u; } v; v.f = f;
  unsigned int u = v.u;
  unsigned int r = u + 0x7FFFu + ((u >> 16) & 1u);   // round-to-nearest-even
  return (ushort)(r >> 16);
}
__device__ __forceinline__ float sigmoidf_(float x) {
  return 1.0f / (1.0f + __expf(-x));
}
__device__ __forceinline__ float tanhf_(float x) {
  return 2.0f / (1.0f + __expf(-2.0f * x)) - 1.0f;
}
__device__ __forceinline__ v8f wmma_bf16(v16bf a, v16bf b, v8f c) {
  return __builtin_amdgcn_wmma_f32_16x16x32_bf16(false, a, false, b, (short)0, c, false, false);
}

// A fragment: 16x32 bf16 tile, A row-major, rows m0.., k-slice k0..
// lanes 0-15: row=lane, K = k0+0..15 ; lanes 16-31: K = k0+16..31.
__device__ __forceinline__ v16bf load_a_frag(const ushort* A, int lda, int m0, int k0, int lane) {
  int row = m0 + (lane & 15);
  int ka  = k0 + (lane >> 4) * 16;
  const ushort* p = A + row * lda + ka;
  Frag32B f;
  f.q.lo = *(const u32x4*)(p);
  f.q.hi = *(const u32x4*)(p + 8);
  return f.v;
}

// B fragment: 32x16 bf16 tile from TRANSPOSED B: Bt[N][K] row-major (ldb=K).
// lanes 0-15: col=lane, K=k0..k0+15 ; lanes 16-31: K=k0+16..k0+31 (contiguous 32B/lane).
__device__ __forceinline__ v16bf load_b_frag(const ushort* Bt, int ldb, int n0, int k0, int lane) {
  int col = n0 + (lane & 15);
  int kb  = k0 + (lane >> 4) * 16;
  const ushort* p = Bt + col * ldb + kb;
  Frag32B f;
  f.q.lo = *(const u32x4*)(p);
  f.q.hi = *(const u32x4*)(p + 8);
  return f.v;
}

// ---------- kernel 1: f32 -> bf16 copy of x ----------
__global__ void k_cvt_x(const float* __restrict__ x, ushort* __restrict__ xb) {
  int i = (blockIdx.x * blockDim.x + threadIdx.x) * 8;
  f32x4 f0 = *(const f32x4*)(x + i);
  f32x4 f1 = *(const f32x4*)(x + i + 4);
  u32x4 o;
  o.x = (unsigned)f2bf(f0.x) | ((unsigned)f2bf(f0.y) << 16);
  o.y = (unsigned)f2bf(f0.z) | ((unsigned)f2bf(f0.w) << 16);
  o.z = (unsigned)f2bf(f1.x) | ((unsigned)f2bf(f1.y) << 16);
  o.w = (unsigned)f2bf(f1.z) | ((unsigned)f2bf(f1.w) << 16);
  *(u32x4*)(xb + i) = o;
}

// ---------- kernel 2: f32 [K][N] -> bf16 transposed [N][K] ----------
__global__ void k_cvt_transpose(const float* __restrict__ in, ushort* __restrict__ out,
                                int K, int N) {
  int tid = blockIdx.x * blockDim.x + threadIdx.x;
  int k = tid / N, n = tid % N;
  out[n * K + k] = f2bf(in[tid]);
}

// ---------- kernel 3: init state + barrier counters ----------
__global__ void k_init(ushort* __restrict__ h0, float* __restrict__ c,
                       int* __restrict__ bars) {
  int i = blockIdx.x * blockDim.x + threadIdx.x;
  if (i < BB * HH) { h0[i] = 0; c[i] = 0.0f; }
  if (i < SS) bars[i] = 0;
}

// ---------- kernel 4: xW = x(bf16) @ W(bf16,T) -> f32 [M,4H] ----------
// block = 256 thr = 8 waves; each wave: 32x64 output tile, software-pipelined k-loop.
// grid = (M/32, N/512).
__global__ void k_xw_gemm(const ushort* __restrict__ xb, const ushort* __restrict__ Wt,
                          float* __restrict__ xW) {
  const int lane = threadIdx.x & 31;
  const int wave = threadIdx.x >> 5;
  const int m0 = blockIdx.x * 32;
  const int n0 = blockIdx.y * 512 + wave * 64;

  v8f acc[2][4] = {};

  // prologue: k-slice 0
  v16bf a0 = load_a_frag(xb, KK, m0 +  0, 0, lane);
  v16bf a1 = load_a_frag(xb, KK, m0 + 16, 0, lane);
  v16bf b0 = load_b_frag(Wt, KK, n0 +  0, 0, lane);
  v16bf b1 = load_b_frag(Wt, KK, n0 + 16, 0, lane);
  v16bf b2 = load_b_frag(Wt, KK, n0 + 32, 0, lane);
  v16bf b3 = load_b_frag(Wt, KK, n0 + 48, 0, lane);

  for (int k0 = 0; k0 < KK; k0 += 32) {
    // issue next k-slice loads before consuming current fragments
    int kn = (k0 + 32 < KK) ? (k0 + 32) : 0;
    __builtin_prefetch((const void*)(xb + (m0 + (lane & 15)) * KK + kn + 32), 0, 1);
    __builtin_prefetch((const void*)(Wt + (n0 + (lane & 15)) * KK + kn + 32), 0, 1);
    v16bf na0 = load_a_frag(xb, KK, m0 +  0, kn, lane);
    v16bf na1 = load_a_frag(xb, KK, m0 + 16, kn, lane);
    v16bf nb0 = load_b_frag(Wt, KK, n0 +  0, kn, lane);
    v16bf nb1 = load_b_frag(Wt, KK, n0 + 16, kn, lane);
    v16bf nb2 = load_b_frag(Wt, KK, n0 + 32, kn, lane);
    v16bf nb3 = load_b_frag(Wt, KK, n0 + 48, kn, lane);

    acc[0][0] = wmma_bf16(a0, b0, acc[0][0]);
    acc[0][1] = wmma_bf16(a0, b1, acc[0][1]);
    acc[0][2] = wmma_bf16(a0, b2, acc[0][2]);
    acc[0][3] = wmma_bf16(a0, b3, acc[0][3]);
    acc[1][0] = wmma_bf16(a1, b0, acc[1][0]);
    acc[1][1] = wmma_bf16(a1, b1, acc[1][1]);
    acc[1][2] = wmma_bf16(a1, b2, acc[1][2]);
    acc[1][3] = wmma_bf16(a1, b3, acc[1][3]);

    a0 = na0; a1 = na1; b0 = nb0; b1 = nb1; b2 = nb2; b3 = nb3;
  }

  const int nn = n0 + (lane & 15);
  const int mofs = 8 * (lane >> 4);
  for (int mt = 0; mt < 2; ++mt) {
    for (int j = 0; j < 8; ++j) {
      long long r = (long long)(m0 + mt * 16 + j + mofs) * G4;
      xW[r + nn +  0] = acc[mt][0][j];
      xW[r + nn + 16] = acc[mt][1][j];
      xW[r + nn + 32] = acc[mt][2][j];
      xW[r + nn + 48] = acc[mt][3][j];
    }
  }
}

// ---------- kernel 5: persistent LSTM recurrence ----------
// 16 blocks x 256 thr = 128 waves. Wave owns 2 batch tiles x 16 h-cols x 4 gates
// (8 accumulators -> 8 independent WMMAs per k-step, halves Ut L2 traffic).
// Software-pipelined k-loop; grid barrier per timestep via per-step counters.
__global__ void k_lstm_recurrence(const float* __restrict__ xW, const ushort* __restrict__ Ut,
                                  ushort* __restrict__ hA, ushort* __restrict__ hB,
                                  float* __restrict__ c,
                                  float* __restrict__ out_hidden,
                                  float* __restrict__ out_hT, float* __restrict__ out_cT,
                                  int* __restrict__ bars) {
  const int lane = threadIdx.x & 31;
  const int wave = threadIdx.x >> 5;
  const int gw = blockIdx.x * 8 + wave;   // 0..127
  const int m0  = (gw & 1) * 32;          // 2 batch tiles: m0, m0+16
  const int n0h = (gw >> 1) * 16;         // h tile
  const int nh  = n0h + (lane & 15);      // this lane's h column
  const int mofs = 8 * (lane >> 4);
  const int nblocks = (int)gridDim.x;

  for (int t = 0; t < SS; ++t) {
    const ushort* hin = (t & 1) ? hB : hA;
    ushort* hout      = (t & 1) ? hA : hB;

    // warm the epilogue's xW rows for this timestep (scattered f32 reads)
    for (int j = 0; j < 4; ++j) {
      long long xr = ((long long)(m0 + j * 8 + mofs) * SS + t) * G4 + nh;
      __builtin_prefetch((const void*)(xW + xr), 0, 1);
      __builtin_prefetch((const void*)(xW + xr + 2 * HH), 0, 1);
    }

    v8f acc[2][4] = {};
    if (t > 0) {  // h_0 == 0: skip GEMM on the first step
      v16bf a0 = load_a_frag(hin, HH, m0 +  0, 0, lane);
      v16bf a1 = load_a_frag(hin, HH, m0 + 16, 0, lane);
      v16bf bi = load_b_frag(Ut, HH, 0 * HH + n0h, 0, lane);
      v16bf bf = load_b_frag(Ut, HH, 1 * HH + n0h, 0, lane);
      v16bf bg = load_b_frag(Ut, HH, 2 * HH + n0h, 0, lane);
      v16bf bo = load_b_frag(Ut, HH, 3 * HH + n0h, 0, lane);

      for (int k0 = 0; k0 < KK; k0 += 32) {
        int kn = (k0 + 32 < KK) ? (k0 + 32) : 0;
        v16bf na0 = load_a_frag(hin, HH, m0 +  0, kn, lane);
        v16bf na1 = load_a_frag(hin, HH, m0 + 16, kn, lane);
        v16bf nbi = load_b_frag(Ut, HH, 0 * HH + n0h, kn, lane);
        v16bf nbf = load_b_frag(Ut, HH, 1 * HH + n0h, kn, lane);
        v16bf nbg = load_b_frag(Ut, HH, 2 * HH + n0h, kn, lane);
        v16bf nbo = load_b_frag(Ut, HH, 3 * HH + n0h, kn, lane);

        acc[0][0] = wmma_bf16(a0, bi, acc[0][0]);
        acc[0][1] = wmma_bf16(a0, bf, acc[0][1]);
        acc[0][2] = wmma_bf16(a0, bg, acc[0][2]);
        acc[0][3] = wmma_bf16(a0, bo, acc[0][3]);
        acc[1][0] = wmma_bf16(a1, bi, acc[1][0]);
        acc[1][1] = wmma_bf16(a1, bf, acc[1][1]);
        acc[1][2] = wmma_bf16(a1, bg, acc[1][2]);
        acc[1][3] = wmma_bf16(a1, bo, acc[1][3]);

        a0 = na0; a1 = na1; bi = nbi; bf = nbf; bg = nbg; bo = nbo;
      }
    }

    for (int mt = 0; mt < 2; ++mt) {
      for (int j = 0; j < 8; ++j) {
        int b = m0 + mt * 16 + j + mofs;
        long long xidx = ((long long)b * SS + t) * G4 + nh;
        float gi = acc[mt][0][j] + xW[xidx + 0 * HH];
        float gf = acc[mt][1][j] + xW[xidx + 1 * HH];
        float gg = acc[mt][2][j] + xW[xidx + 2 * HH];
        float go = acc[mt][3][j] + xW[xidx + 3 * HH];
        float it = sigmoidf_(gi);
        float ft = sigmoidf_(gf);
        float gt = tanhf_(gg);
        float ot = sigmoidf_(go);
        int cidx = b * HH + nh;
        float cn = ft * c[cidx] + it * gt;
        float hn = ot * tanhf_(cn);
        c[cidx]    = cn;
        hout[cidx] = f2bf(hn);
        out_hidden[((long long)b * SS + t) * HH + nh] = hn;
        if (t == SS - 1) { out_hT[cidx] = hn; out_cT[cidx] = cn; }
      }
    }

    // ---- grid-wide barrier (16 co-resident blocks) ----
    __threadfence();
    __syncthreads();
    if (threadIdx.x == 0) {
      __hip_atomic_fetch_add(&bars[t], 1, __ATOMIC_RELEASE, __HIP_MEMORY_SCOPE_AGENT);
      while (__hip_atomic_load(&bars[t], __ATOMIC_ACQUIRE, __HIP_MEMORY_SCOPE_AGENT) < nblocks) {
        __builtin_amdgcn_s_sleep(1);
      }
    }
    __syncthreads();
  }
}

// ---------- launcher ----------
extern "C" void kernel_launch(void* const* d_in, const int* in_sizes, int n_in,
                              void* d_out, int out_size, void* d_ws, size_t ws_size,
                              hipStream_t stream) {
  (void)in_sizes; (void)n_in; (void)out_size; (void)ws_size;
  const float* x = (const float*)d_in[0];   // [64,512,1024]
  const float* W = (const float*)d_in[1];   // [1024,4096]
  const float* U = (const float*)d_in[2];   // [1024,4096]

  char* ws = (char*)d_ws;
  size_t off = 0;
  ushort* xb  = (ushort*)(ws + off); off += (size_t)MM * II * 2;      //  64 MB
  ushort* Wt  = (ushort*)(ws + off); off += (size_t)G4 * II * 2;      //   8 MB
  ushort* Ut  = (ushort*)(ws + off); off += (size_t)G4 * HH * 2;      //   8 MB
  float*  xW  = (float*) (ws + off); off += (size_t)MM * G4 * 4;      // 512 MB
  ushort* hA  = (ushort*)(ws + off); off += (size_t)BB * HH * 2;
  ushort* hB  = (ushort*)(ws + off); off += (size_t)BB * HH * 2;
  float*  cSt = (float*) (ws + off); off += (size_t)BB * HH * 4;
  int*    bar = (int*)   (ws + off); off += (size_t)SS * 4;

  float* out_hidden = (float*)d_out;                          // [B,S,H]
  float* out_hT = out_hidden + (size_t)BB * SS * HH;          // [B,H]
  float* out_cT = out_hT + (size_t)BB * HH;                   // [B,H]

  // precision conversion + transposes
  k_cvt_x<<<(MM * II) / (8 * 256), 256, 0, stream>>>(x, xb);
  k_cvt_transpose<<<(II * G4) / 256, 256, 0, stream>>>(W, Wt, II, G4);
  k_cvt_transpose<<<(HH * G4) / 256, 256, 0, stream>>>(U, Ut, HH, G4);
  k_init<<<(BB * HH) / 256, 256, 0, stream>>>(hA, cSt, bar);

  // xW = x @ W  (bf16 WMMA, f32 accum)
  k_xw_gemm<<<dim3(MM / 32, G4 / 512), 256, 0, stream>>>(xb, Wt, xW);

  // serial recurrence, one persistent kernel
  k_lstm_recurrence<<<16, 256, 0, stream>>>(xW, Ut, hA, hB, cSt,
                                            out_hidden, out_hT, out_cT, bar);
}
